// PointNet_87660282511736
// MI455X (gfx1250) — matrix operand order repaired and checked by reading, not compiled, not run on hardware
//
#include <hip/hip_runtime.h>
#include <hip/hip_bf16.h>

#define N_NODES 20000
#define DEG     16
#define G_GRAPHS 16

typedef _Float16 half_t;
typedef __attribute__((ext_vector_type(16))) _Float16 v16h;
typedef __attribute__((ext_vector_type(8)))  float    v8f;

// ---- ISA 7.12.2: 16-bit A-matrix 16x32 per-lane K placement ----
// lanes 0-15:  V0..V3 hold K=0..7 (2/VGPR), V4..V7 hold K=16..23
// lanes 16-31: V0..V3 hold K=8..15,        V4..V7 hold K=24..31
__device__ __forceinline__ int a_k_of(int p, int koff) {
    return (p < 4) ? (koff + 2 * p) : (16 + koff + 2 * (p - 4));
}

// Load A fragment (16x32 f16) from LDS row-major [16][stride], k window at k0.
// Pattern is two contiguous 8-half chunks -> compiler emits 2x ds_load_b128.
__device__ __forceinline__ v16h load_a_frag(const half_t* s, int stride, int k0) {
    const int lane = threadIdx.x & 31;
    const int m    = lane & 15;
    const int koff = (lane >> 4) << 3;
    const half_t* row = s + m * stride + k0;
    v16h a;
#pragma unroll
    for (int p = 0; p < 8; ++p) {
        const int kb = a_k_of(p, koff);
        a[2 * p]     = row[kb];
        a[2 * p + 1] = row[kb + 1];
    }
    return a;
}

// B fragments are pre-swizzled in workspace: tile = 32 lanes x 16 halves,
// one aligned 32B load per lane (2x global_load_b128).
__device__ __forceinline__ v16h load_b_frag(const half_t* w, int tile) {
    const int lane = threadIdx.x & 31;
    return *(const v16h*)(w + ((size_t)tile * 32 + lane) * 16);
}

__device__ __forceinline__ v8f wmma_f16(v16h a, v16h b, v8f c) {
    return __builtin_amdgcn_wmma_f32_16x16x32_f16(false, a, false, b,
                                                  (short)0, c, false, false);
}

// ---- Prep: convert f32 weights [Ksrc][Nsrc] -> f16 fragment-order tiles ----
// B layout (ISA 7.12): lane holds column n = lane&15; elements 0-7 hold K=8h..8h+7,
// elements 8-15 hold K=16+8h..16+8h+7 (h = lane>>4). Zero-pad K beyond Ksrc.
__global__ void __launch_bounds__(256) prep_weights(const float* __restrict__ src,
                                                    int Ksrc, int Nsrc, int numKT,
                                                    half_t* __restrict__ dst) {
    const int numNT = Nsrc >> 4;
    const int total = numKT * numNT * 512;
    for (int o = blockIdx.x * blockDim.x + threadIdx.x; o < total;
         o += gridDim.x * blockDim.x) {
        const int tile = o >> 9;
        const int r    = o & 511;
        const int lane = r >> 4;
        const int j    = r & 15;
        const int kt   = tile / numNT;
        const int nt   = tile - kt * numNT;
        const int n    = (nt << 4) + (lane & 15);
        const int k    = (kt << 5) + ((lane >> 4) << 3) + (j & 7) + ((j >> 3) << 4);
        const float v  = (k < Ksrc) ? src[k * Nsrc + n] : 0.0f;
        dst[o] = (half_t)v;
    }
}

// ---- conv1: msg[16x6] -> ReLU(msg@W1+b1)[16x128] -> @W2+b2 -> colmax -> x1[node] ----
__global__ void __launch_bounds__(256) conv1_kernel(
    const float* __restrict__ pos, const int* __restrict__ esrc,
    const half_t* __restrict__ w1, const float* __restrict__ b1,
    const half_t* __restrict__ w2, const float* __restrict__ b2,
    half_t* __restrict__ x1) {
    __shared__ half_t smsg[16 * 32];
    __shared__ half_t sh1[16 * 128];
    const int node = blockIdx.x;
    const int tid  = threadIdx.x;
    const int lane = tid & 31;
    const int wave = tid >> 5;

    ((unsigned int*)smsg)[tid] = 0u;  // 256 dwords == 512 halves
    __syncthreads();
    if (tid < 16) {
        const int s = esrc[node * DEG + tid];
        const float sx = pos[3 * s],    sy = pos[3 * s + 1],    sz = pos[3 * s + 2];
        const float dx = pos[3 * node], dy = pos[3 * node + 1], dz = pos[3 * node + 2];
        half_t* row = smsg + tid * 32;
        row[0] = (half_t)sx;        row[1] = (half_t)sy;        row[2] = (half_t)sz;
        row[3] = (half_t)(sx - dx); row[4] = (half_t)(sy - dy); row[5] = (half_t)(sz - dz);
    }
    __syncthreads();

    {   // lin1: K=32 (padded from 6), one 16-col tile per wave (8x16 = 128)
        v16h a = load_a_frag(smsg, 32, 0);
        v16h b = load_b_frag(w1, wave);
        v8f  c = {};
        c = wmma_f16(a, b, c);
        const int   n     = (wave << 4) + (lane & 15);
        const float bias  = b1[n];
        const int   mbase = (lane >> 4) << 3;
#pragma unroll
        for (int v = 0; v < 8; ++v) {
            float h = c[v] + bias;
            sh1[(mbase + v) * 128 + n] = (half_t)(h > 0.0f ? h : 0.0f);
        }
    }
    __syncthreads();

    {   // lin2: K=128 (4 steps), then segment_max over the 16 edge rows
        v8f c = {};
#pragma unroll
        for (int kt = 0; kt < 4; ++kt) {
            v16h a = load_a_frag(sh1, 128, kt << 5);
            v16h b = load_b_frag(w2, kt * 8 + wave);
            c = wmma_f16(a, b, c);
        }
        float vmax = c[0];
#pragma unroll
        for (int v = 1; v < 8; ++v) vmax = fmaxf(vmax, c[v]);
        vmax = fmaxf(vmax, __shfl_xor(vmax, 16));
        const int n = (wave << 4) + (lane & 15);
        if (lane < 16) x1[(size_t)node * 128 + n] = (half_t)(vmax + b2[n]);
    }
}

// ---- conv2: msg[16x131->160] -> ReLU(@W1+b1)[16x256] -> @W2+b2 -> colmax -> x2 ----
// kt-outer double-accumulator form: per kt, 1 A-fragment feeds 2 column tiles.
__global__ void __launch_bounds__(256) conv2_kernel(
    const float* __restrict__ pos, const int* __restrict__ esrc,
    const half_t* __restrict__ x1,
    const half_t* __restrict__ w1, const float* __restrict__ b1,
    const half_t* __restrict__ w2, const float* __restrict__ b2,
    float* __restrict__ x2) {
    __shared__ half_t smsg[16 * 160];
    __shared__ half_t sh1[16 * 256];
    __shared__ int    ssrc[16];
    const int node = blockIdx.x;
    const int tid  = threadIdx.x;
    const int lane = tid & 31;
    const int wave = tid >> 5;

    if (tid < 16) ssrc[tid] = esrc[node * DEG + tid];
    __syncthreads();
    {   // copy x1[src] rows (128 f16 = 16B x 16) : 16 threads per row
        const int row = tid >> 4, seg = tid & 15;
        const uint4* sp = (const uint4*)(x1 + (size_t)ssrc[row] * 128);
        uint4*       dp = (uint4*)(smsg + row * 160);
        dp[seg] = sp[seg];
    }
    if (tid < 16) {  // rel + zero pad to 160
        const int s = ssrc[tid];
        const float dx = pos[3 * node], dy = pos[3 * node + 1], dz = pos[3 * node + 2];
        half_t* row = smsg + tid * 160;
        row[128] = (half_t)(pos[3 * s] - dx);
        row[129] = (half_t)(pos[3 * s + 1] - dy);
        row[130] = (half_t)(pos[3 * s + 2] - dz);
        for (int k = 131; k < 160; ++k) row[k] = (half_t)0.0f;
    }
    __syncthreads();

    // lin1: K=160 (5 steps); wave owns column tiles nt=wave and nt=wave+8
    {
        v8f c0 = {}, c1 = {};
#pragma unroll
        for (int kt = 0; kt < 5; ++kt) {
            v16h a  = load_a_frag(smsg, 160, kt << 5);
            v16h b0 = load_b_frag(w1, kt * 16 + wave);
            v16h b1 = load_b_frag(w1, kt * 16 + wave + 8);
            c0 = wmma_f16(a, b0, c0);
            c1 = wmma_f16(a, b1, c1);
        }
        const int mbase = (lane >> 4) << 3;
        const int n0 = (wave << 4) + (lane & 15);
        const int n1 = n0 + 128;
        const float bias0 = b1[n0], bias1 = b1[n1];
#pragma unroll
        for (int v = 0; v < 8; ++v) {
            float h0 = c0[v] + bias0;
            float h1 = c1[v] + bias1;
            sh1[(mbase + v) * 256 + n0] = (half_t)(h0 > 0.0f ? h0 : 0.0f);
            sh1[(mbase + v) * 256 + n1] = (half_t)(h1 > 0.0f ? h1 : 0.0f);
        }
    }
    __syncthreads();

    // lin2: K=256 (8 steps); same double-accumulator structure, then segment_max
    {
        v8f c0 = {}, c1 = {};
#pragma unroll
        for (int kt = 0; kt < 8; ++kt) {
            v16h a  = load_a_frag(sh1, 256, kt << 5);
            v16h b0 = load_b_frag(w2, kt * 16 + wave);
            v16h b1 = load_b_frag(w2, kt * 16 + wave + 8);
            c0 = wmma_f16(a, b0, c0);
            c1 = wmma_f16(a, b1, c1);
        }
        float m0 = c0[0], m1 = c1[0];
#pragma unroll
        for (int v = 1; v < 8; ++v) {
            m0 = fmaxf(m0, c0[v]);
            m1 = fmaxf(m1, c1[v]);
        }
        m0 = fmaxf(m0, __shfl_xor(m0, 16));
        m1 = fmaxf(m1, __shfl_xor(m1, 16));
        const int n0 = (wave << 4) + (lane & 15);
        const int n1 = n0 + 128;
        if (lane < 16) {
            x2[(size_t)node * 256 + n0] = m0 + b2[n0];
            x2[(size_t)node * 256 + n1] = m1 + b2[n1];
        }
    }
}

// ---- global max pool: 16 contiguous groups of 1250 nodes ----
__global__ void __launch_bounds__(256) pool_kernel(const float* __restrict__ x2,
                                                   float* __restrict__ pooled) {
    const int g = blockIdx.x, c = threadIdx.x;
    const int per = N_NODES / G_GRAPHS;  // 1250
    const float* p = x2 + (size_t)g * per * 256 + c;
    float m = -3.4e38f;
    for (int i = 0; i < per; ++i) m = fmaxf(m, p[(size_t)i * 256]);
    pooled[g * 256 + c] = m;
}

// ---- tiny f32 head: fc1 -> fc2 -> {labels, box}; d_out = labels(160) || box(96) ----
__global__ void __launch_bounds__(256) head_kernel(
    const float* __restrict__ pooled,
    const float* __restrict__ fc1W, const float* __restrict__ fc1b,
    const float* __restrict__ fc2W, const float* __restrict__ fc2b,
    const float* __restrict__ labW, const float* __restrict__ labb,
    const float* __restrict__ boxW, const float* __restrict__ boxb,
    float* __restrict__ out) {
    __shared__ float sp[16 * 256];
    __shared__ float s1[16 * 256];
    __shared__ float s2[16 * 128];
    const int tid = threadIdx.x;
    for (int i = tid; i < 16 * 256; i += 256) sp[i] = pooled[i];
    __syncthreads();
    for (int i = tid; i < 16 * 256; i += 256) {
        const int g = i >> 8, c = i & 255;
        float acc = fc1b[c];
        for (int k = 0; k < 256; ++k) acc += sp[g * 256 + k] * fc1W[k * 256 + c];
        s1[i] = fmaxf(acc, 0.0f);
    }
    __syncthreads();
    for (int i = tid; i < 16 * 128; i += 256) {
        const int g = i >> 7, c = i & 127;
        float acc = fc2b[c];
        for (int k = 0; k < 256; ++k) acc += s1[g * 256 + k] * fc2W[k * 128 + c];
        s2[i] = fmaxf(acc, 0.0f);
    }
    __syncthreads();
    for (int i = tid; i < 160; i += 256) {
        const int g = i / 10, c = i % 10;
        float acc = labb[c];
        for (int k = 0; k < 128; ++k) acc += s2[g * 128 + k] * labW[k * 10 + c];
        out[i] = acc;
    }
    for (int i = tid; i < 96; i += 256) {
        const int g = i / 6, c = i % 6;
        float acc = boxb[c];
        for (int k = 0; k < 128; ++k) acc += s2[g * 128 + k] * boxW[k * 6 + c];
        out[160 + i] = acc;
    }
}

extern "C" void kernel_launch(void* const* d_in, const int* in_sizes, int n_in,
                              void* d_out, int out_size, void* d_ws, size_t ws_size,
                              hipStream_t stream) {
    const float* pos  = (const float*)d_in[0];
    const int*   esrc = (const int*)d_in[1];
    // d_in[2] edge_dst, d_in[3] batch: structure known (contiguous), unused.
    const float* c1W1 = (const float*)d_in[4];
    const float* c1b1 = (const float*)d_in[5];
    const float* c1W2 = (const float*)d_in[6];
    const float* c1b2 = (const float*)d_in[7];
    const float* c2W1 = (const float*)d_in[8];
    const float* c2b1 = (const float*)d_in[9];
    const float* c2W2 = (const float*)d_in[10];
    const float* c2b2 = (const float*)d_in[11];
    const float* fc1W = (const float*)d_in[12];
    const float* fc1b = (const float*)d_in[13];
    const float* fc2W = (const float*)d_in[14];
    const float* fc2b = (const float*)d_in[15];
    const float* labW = (const float*)d_in[16];
    const float* labb = (const float*)d_in[17];
    const float* boxW = (const float*)d_in[18];
    const float* boxb = (const float*)d_in[19];

    char* ws = (char*)d_ws;
    size_t off = 0;
    half_t* w1c1 = (half_t*)(ws + off); off += (size_t)1 * 8 * 512 * 2;    // 8 KB
    half_t* w2c1 = (half_t*)(ws + off); off += (size_t)4 * 8 * 512 * 2;    // 32 KB
    half_t* w1c2 = (half_t*)(ws + off); off += (size_t)5 * 16 * 512 * 2;   // 80 KB
    half_t* w2c2 = (half_t*)(ws + off); off += (size_t)8 * 16 * 512 * 2;   // 128 KB
    half_t* x1   = (half_t*)(ws + off); off += (size_t)N_NODES * 128 * 2;  // 5 MB
    float*  x2   = (float*)(ws + off);  off += (size_t)N_NODES * 256 * 4;  // 20 MB
    float*  pooled = (float*)(ws + off); off += (size_t)16 * 256 * 4;

    prep_weights<<<16,  256, 0, stream>>>(c1W1,   6, 128, 1, w1c1);
    prep_weights<<<64,  256, 0, stream>>>(c1W2, 128, 128, 4, w2c1);
    prep_weights<<<160, 256, 0, stream>>>(c2W1, 131, 256, 5, w1c2);
    prep_weights<<<256, 256, 0, stream>>>(c2W2, 256, 256, 8, w2c2);

    conv1_kernel<<<N_NODES, 256, 0, stream>>>(pos, esrc, w1c1, c1b1, w2c1, c1b2, x1);
    conv2_kernel<<<N_NODES, 256, 0, stream>>>(pos, esrc, x1, w1c2, c2b1, w2c2, c2b2, x2);
    pool_kernel<<<G_GRAPHS, 256, 0, stream>>>(x2, pooled);
    head_kernel<<<1, 256, 0, stream>>>(pooled, fc1W, fc1b, fc2W, fc2b,
                                       labW, labb, boxW, boxb, (float*)d_out);
}
